// MultilinearAttention_71116068487371
// MI455X (gfx1250) — matrix operand order repaired and checked by reading, not compiled, not run on hardware
//
#include <hip/hip_runtime.h>

// ---------------------------------------------------------------------------
// MultilinearAttention forward for gfx1250 (MI455X), bf16 WMMA pipeline.
// GEMM tiles staged by the Tensor Data Mover (tensor_load_to_lds, TENSORcnt)
// with async global->LDS (ASYNCcnt) as fallback / secondary path.
// ---------------------------------------------------------------------------

typedef __bf16 v16bf __attribute__((ext_vector_type(16)));
typedef float  v8f   __attribute__((ext_vector_type(8)));
typedef unsigned int u32x4 __attribute__((ext_vector_type(4)));
typedef int          i32x4 __attribute__((ext_vector_type(4)));
typedef int          i32x8 __attribute__((ext_vector_type(8)));

struct __align__(16) U128 { unsigned int x, y, z, w; };

union Frag { v16bf v; U128 q[2]; };
union U4S  { U128 q; unsigned short s[8]; };

#if defined(__HIP_DEVICE_COMPILE__) && __has_builtin(__builtin_amdgcn_tensor_load_to_lds)
#define HAVE_TDM 1
#else
#define HAVE_TDM 0
#endif

__device__ __forceinline__ unsigned short f2bf(float f) {
    unsigned u = __float_as_uint(f);
    u += 0x7FFFu + ((u >> 16) & 1u);   // round-to-nearest-even
    return (unsigned short)(u >> 16);
}

// CDNA5 async DMA: global -> LDS, 16B per lane, tracked by ASYNCcnt.
__device__ __forceinline__ void async_ld128(const unsigned short* g,
                                            const unsigned short* l) {
    unsigned lds_off = (unsigned)(unsigned long long)(uintptr_t)l;
    asm volatile("global_load_async_to_lds_b128 %0, %1, off"
                 :
                 : "v"(lds_off), "v"(g)
                 : "memory");
}
__device__ __forceinline__ void wait_async0() {
    asm volatile("s_wait_asynccnt 0x0" ::: "memory");
}

#if HAVE_TDM
// TDM 2D tile load: tile_dim0 (K) x tile_dim1 (rows) described by g1,
// global tile start + LDS destination in g0. One instruction moves the tile.
// This toolchain declares the 6-arg builtin: (g0, g1, g2, g3, g4, cpol).
__device__ __forceinline__ void tdm_load_2d(const void* gptr, unsigned lds_off,
                                            i32x8 g1) {
    unsigned long long ga = (unsigned long long)(uintptr_t)gptr;
    u32x4 g0;
    g0[0] = 1u;                                   // count=1, user descriptor
    g0[1] = lds_off;                              // lds_addr [63:32]
    g0[2] = (unsigned)ga;                         // global_addr [95:64]
    g0[3] = (unsigned)(ga >> 32) | (2u << 30);    // addr[56:32] | type=2
    i32x4 z4 = { 0, 0, 0, 0 };                    // groups 2/3 unused (2D tile)
    i32x8 z8 = { 0, 0, 0, 0, 0, 0, 0, 0 };
    __builtin_amdgcn_tensor_load_to_lds(g0, g1, z4, z4, z8, 0);
}
// Group1 for: bf16 (2B), tile 32 x 128, tensor row stride = kt elements,
// LDS padding 48B after every 64B (-> 112B row stride == LDP*2).
__device__ __forceinline__ i32x8 tdm_group1(unsigned kt) {
    const unsigned DIM1 = 1u << 24;               // rows in tensor (no clipping)
    i32x8 g1;
    g1[0] = (int)((1u << 16)        // data_size = 2 bytes
                | (1u << 20)        // pad_enable
                | (3u << 22)        // pad_interval: 16 DWORDs (64B)
                | (11u << 25));     // pad_amount: 12 DWORDs (48B)
    g1[1] = (int)((kt & 0xFFFFu) << 16);                         // dim0[15:0]
    g1[2] = (int)(((kt >> 16) & 0xFFFFu) | ((DIM1 & 0xFFFFu) << 16));
    g1[3] = (int)(((DIM1 >> 16) & 0xFFFFu) | (32u << 16));       // tile_dim0=32
    g1[4] = (int)128u;                                           // tile_dim1=128
    g1[5] = (int)kt;                                             // dim0_stride lo
    g1[6] = 0;
    g1[7] = 0;
    return g1;
}
#endif

// ------------------------------ constants ----------------------------------
#define NB_ROWS 16384u        // B*N
#define CC      1024u         // channels
#define LDP     56            // padded LDS row stride (elements): 112B, 16B-aligned, conflict-free

// workspace layout (bytes)
#define SZ_MAT   ((size_t)NB_ROWS * CC * 2)        // 32 MB bf16 activation matrix
#define OFF_XB   ((size_t)0)
#define OFF_QB   (OFF_XB  + SZ_MAT)
#define OFF_K0B  (OFF_QB  + SZ_MAT)
#define OFF_K1B  (OFF_K0B + SZ_MAT)
#define OFF_V0B  (OFF_K1B + SZ_MAT)
#define OFF_V1B  (OFF_V0B + SZ_MAT)
#define OFF_WQB  (OFF_V1B + SZ_MAT)
#define OFF_WKB  (OFF_WQB + (size_t)CC * CC * 2)
#define OFF_WVB  (OFF_WKB + (size_t)2 * CC * CC * 2)
#define OFF_WOB  (OFF_WVB + (size_t)2 * CC * CC * 2)
#define OFF_ST   (OFF_WOB + (size_t)CC * CC * 2)           // stateT: 64*(64*64) bf16
#define OFF_OP   (OFF_ST  + (size_t)64 * 64 * 64 * 2)      // out_pre: f32 [16384][1024]
#define OFF_XLN  (OFF_OP  + (size_t)NB_ROWS * CC * 4)      // xln bf16

// ------------------------------ convert ------------------------------------
__global__ __launch_bounds__(256) void cvt_bf16_kernel(
    const float* __restrict__ src, unsigned short* __restrict__ dst, int n) {
    int i = blockIdx.x * 256 + threadIdx.x;
    if (i < n) dst[i] = f2bf(src[i]);
}

// ------------------------------ generic WMMA GEMM --------------------------
// C[M,Nout] = A[M,K] * Bt[Nout,K]^T  (+ epilogue)
//   mode 0: q  -> +resid +bias, silu+0.5, per-head(64) L2 norm, bf16 out
//   mode 1: k  -> same as q, then *scale, bf16 out
//   mode 2: v  -> +resid +bias, *scale, bf16 out
//   mode 3: Wo -> +bias, f32 out
// Double-buffered LDS tiles filled by the TDM (or async DMA fallback).
__global__ __launch_bounds__(256) void gemm_bf16_kernel(
    const unsigned short* __restrict__ A, const unsigned short* __restrict__ Bt,
    const float* __restrict__ resid, const float* __restrict__ bias,
    unsigned short* __restrict__ outb, float* __restrict__ outf,
    int Kt, int Nout, int mode, float scale)
{
    __shared__ unsigned short As[2][128 * LDP];
    __shared__ unsigned short Bs[2][128 * LDP];

    const int t    = threadIdx.x;
    const int lane = t & 31, wave = t >> 5;
    const int half = lane >> 4, ln16 = lane & 15;
    const int wm = wave & 3, wn = wave >> 2;
    const size_t m0 = (size_t)blockIdx.x * 128;
    const size_t n0 = (size_t)blockIdx.y * 128;

    v8f acc[2][4] = {};

#if HAVE_TDM
    const i32x8 g1 = tdm_group1((unsigned)Kt);
    // wave 0 drives the Tensor Data Mover for both tiles
    auto stage = [&](int buf, int k0) {
        if (t < 32) {
            tdm_load_2d(A  + m0 * (size_t)Kt + k0, (unsigned)(uintptr_t)&As[buf][0], g1);
            tdm_load_2d(Bt + n0 * (size_t)Kt + k0, (unsigned)(uintptr_t)&Bs[buf][0], g1);
        }
    };
    auto wait_dma = [&]() { if (t < 32) __builtin_amdgcn_s_wait_tensorcnt((short)0); };
#else
    auto stage = [&](int buf, int k0) {
#pragma unroll
        for (int i = 0; i < 2; ++i) {
            int c = t + i * 256;
            int row = c >> 2, ch = (c & 3) * 8;
            async_ld128(A  + (m0 + row) * (size_t)Kt + k0 + ch, &As[buf][row * LDP + ch]);
            async_ld128(Bt + (n0 + row) * (size_t)Kt + k0 + ch, &Bs[buf][row * LDP + ch]);
        }
    };
    auto wait_dma = [&]() { wait_async0(); };
#endif

    stage(0, 0);
    wait_dma();
    __syncthreads();

    int cur = 0;
    for (int k0 = 0; k0 < Kt; k0 += 32) {
        if (k0 + 32 < Kt) {
            stage(cur ^ 1, k0 + 32);                      // overlap DMA with compute
            if (k0 + 64 < Kt)                             // L2 warm-up two tiles ahead
                __builtin_prefetch(A + (m0 + (t >> 1)) * (size_t)Kt + k0 + 64, 0, 1);
        }

        Frag a[2], b[4];
#pragma unroll
        for (int mt = 0; mt < 2; ++mt) {
            const unsigned short* p = &As[cur][(wm * 32 + mt * 16 + ln16) * LDP];
            a[mt].q[0] = *reinterpret_cast<const U128*>(p + half * 8);
            a[mt].q[1] = *reinterpret_cast<const U128*>(p + 16 + half * 8);
        }
#pragma unroll
        for (int nt = 0; nt < 4; ++nt) {
            const unsigned short* p = &Bs[cur][(wn * 64 + nt * 16 + ln16) * LDP];
            b[nt].q[0] = *reinterpret_cast<const U128*>(p + half * 8);
            b[nt].q[1] = *reinterpret_cast<const U128*>(p + 16 + half * 8);
        }
#pragma unroll
        for (int mt = 0; mt < 2; ++mt)
#pragma unroll
            for (int nt = 0; nt < 4; ++nt)
                acc[mt][nt] = __builtin_amdgcn_wmma_f32_16x16x32_bf16(
                    false, a[mt].v, false, b[nt].v, (short)0, acc[mt][nt], false, false);

        wait_dma();           // next tile has landed in LDS
        __syncthreads();      // everyone done reading `cur`, next tile visible
        cur ^= 1;
    }

    // epilogue
#pragma unroll
    for (int mt = 0; mt < 2; ++mt) {
#pragma unroll
        for (int r = 0; r < 8; ++r) {
            size_t row = m0 + wm * 32 + mt * 16 + half * 8 + r;
            float y[4];
#pragma unroll
            for (int nt = 0; nt < 4; ++nt) {
                size_t col = n0 + wn * 64 + nt * 16 + ln16;
                float v = acc[mt][nt][r];
                if (resid) v += resid[row * (size_t)Nout + col];
                v += bias[col];
                if (mode <= 1) v = v * (1.0f / (1.0f + __expf(-v))) + 0.5f;  // silu + 0.5
                y[nt] = v;
            }
            if (mode <= 1) {
                // per-head L2 norm: this wave's 64 columns == one head
                float ss = y[0]*y[0] + y[1]*y[1] + y[2]*y[2] + y[3]*y[3];
                ss += __shfl_xor(ss, 1, 32);
                ss += __shfl_xor(ss, 2, 32);
                ss += __shfl_xor(ss, 4, 32);
                ss += __shfl_xor(ss, 8, 32);
                float rn = 1.0f / (sqrtf(ss) + 1e-6f);
#pragma unroll
                for (int nt = 0; nt < 4; ++nt) y[nt] *= rn;
            }
#pragma unroll
            for (int nt = 0; nt < 4; ++nt) {
                size_t col = n0 + wn * 64 + nt * 16 + ln16;
                float v = y[nt] * scale;
                if (mode == 3) outf[row * (size_t)Nout + col] = v;
                else           outb[row * (size_t)Nout + col] = f2bf(v);
            }
        }
    }
}

// ------------------------------ state kernel -------------------------------
// per (b,h): S_k[d,e] = sum_n ks[n,d]*vs[n,e] for k=0,1 ; state=S0*S1 (elemwise)
// stored transposed as stateT[e][d] (bf16) so the next GEMM's B-staging is direct.
// (staging transposes during LDS write -> stays on the synchronous path)
__global__ __launch_bounds__(256) void state_kernel(
    const unsigned short* __restrict__ k0b, const unsigned short* __restrict__ k1b,
    const unsigned short* __restrict__ v0b, const unsigned short* __restrict__ v1b,
    unsigned short* __restrict__ stateT)
{
    __shared__ unsigned short Ak[2][64 * LDP];
    __shared__ unsigned short Bv[2][64 * LDP];

    const int bh = blockIdx.x;
    const int b = bh >> 4, h = bh & 15;
    const int t = threadIdx.x;
    const int lane = t & 31, wave = t >> 5;
    const int half = lane >> 4, ln16 = lane & 15;
    const int wm = wave & 3, wn = wave >> 2;
    const int j  = t >> 3;          // n row within chunk (0..31)
    const int d0 = (t & 7) * 8;     // channel base

    const unsigned short* ks[2]  = { k0b, k1b };
    const unsigned short* vsp[2] = { v0b, v1b };

    v8f acc[2][2] = {};

    for (int n0 = 0; n0 < 4096; n0 += 32) {
        size_t src = ((size_t)(b * 4096 + n0 + j)) * CC + h * 64 + d0;
        U4S da[2], db[2];
#pragma unroll
        for (int kk = 0; kk < 2; ++kk) {           // issue all 4 loads first
            da[kk].q = *reinterpret_cast<const U128*>(ks[kk]  + src);
            db[kk].q = *reinterpret_cast<const U128*>(vsp[kk] + src);
        }
#pragma unroll
        for (int kk = 0; kk < 2; ++kk)
#pragma unroll
            for (int i = 0; i < 8; ++i) {          // transpose into [d][n]
                Ak[kk][(d0 + i) * LDP + j] = da[kk].s[i];
                Bv[kk][(d0 + i) * LDP + j] = db[kk].s[i];
            }
        __syncthreads();
#pragma unroll
        for (int kk = 0; kk < 2; ++kk) {
            Frag a, bf;
            const unsigned short* pa = &Ak[kk][(wm * 16 + ln16) * LDP];
            a.q[0] = *reinterpret_cast<const U128*>(pa + half * 8);
            a.q[1] = *reinterpret_cast<const U128*>(pa + 16 + half * 8);
#pragma unroll
            for (int nt = 0; nt < 2; ++nt) {
                const unsigned short* pb = &Bv[kk][(wn * 32 + nt * 16 + ln16) * LDP];
                bf.q[0] = *reinterpret_cast<const U128*>(pb + half * 8);
                bf.q[1] = *reinterpret_cast<const U128*>(pb + 16 + half * 8);
                acc[kk][nt] = __builtin_amdgcn_wmma_f32_16x16x32_bf16(
                    false, a.v, false, bf.v, (short)0, acc[kk][nt], false, false);
            }
        }
        __syncthreads();
    }
#pragma unroll
    for (int nt = 0; nt < 2; ++nt)
#pragma unroll
        for (int r = 0; r < 8; ++r) {
            int d = wm * 16 + half * 8 + r;
            int e = wn * 32 + nt * 16 + ln16;
            float v = acc[0][nt][r] * acc[1][nt][r];
            stateT[(size_t)bh * 4096 + e * 64 + d] = f2bf(v);
        }
}

// ------------------------------ out kernel ---------------------------------
// per (b,h): out_pre[b,n,h*64+e] = sum_d q[n,d] * state[d,e]  (f32 out, pre-LN)
// Both K tiles staged via async global->LDS; single wait + barrier.
__global__ __launch_bounds__(256) void out_kernel(
    const unsigned short* __restrict__ qb, const unsigned short* __restrict__ stateT,
    float* __restrict__ out_pre)
{
    __shared__ unsigned short As[2][128 * LDP];
    __shared__ unsigned short Bs[2][64 * LDP];

    const int bh = blockIdx.y;
    const int b = bh >> 4, h = bh & 15;
    const size_t n0 = (size_t)blockIdx.x * 128;
    const int t = threadIdx.x;
    const int lane = t & 31, wave = t >> 5;
    const int half = lane >> 4, ln16 = lane & 15;

    v8f acc[4] = {};

#pragma unroll
    for (int kb = 0; kb < 2; ++kb) {
        const int k0 = kb * 32;
#pragma unroll
        for (int i = 0; i < 2; ++i) {
            int c = t + i * 256;
            int row = c >> 2, ch = (c & 3) * 8;
            async_ld128(qb + ((size_t)b * 4096 + n0 + row) * CC + h * 64 + k0 + ch,
                        &As[kb][row * LDP + ch]);
        }
        {
            int row = t >> 2, ch = (t & 3) * 8;
            async_ld128(stateT + (size_t)bh * 4096 + row * 64 + k0 + ch,
                        &Bs[kb][row * LDP + ch]);
        }
    }
    wait_async0();
    __syncthreads();

#pragma unroll
    for (int kb = 0; kb < 2; ++kb) {
        Frag a, bf;
        const unsigned short* pa = &As[kb][(wave * 16 + ln16) * LDP];
        a.q[0] = *reinterpret_cast<const U128*>(pa + half * 8);
        a.q[1] = *reinterpret_cast<const U128*>(pa + 16 + half * 8);
#pragma unroll
        for (int nt = 0; nt < 4; ++nt) {
            const unsigned short* pb = &Bs[kb][(nt * 16 + ln16) * LDP];
            bf.q[0] = *reinterpret_cast<const U128*>(pb + half * 8);
            bf.q[1] = *reinterpret_cast<const U128*>(pb + 16 + half * 8);
            acc[nt] = __builtin_amdgcn_wmma_f32_16x16x32_bf16(
                false, a.v, false, bf.v, (short)0, acc[nt], false, false);
        }
    }
#pragma unroll
    for (int nt = 0; nt < 4; ++nt)
#pragma unroll
        for (int r = 0; r < 8; ++r) {
            size_t n = n0 + wave * 16 + half * 8 + r;
            int e = nt * 16 + ln16;
            out_pre[((size_t)b * 4096 + n) * CC + h * 64 + e] = acc[nt][r];
        }
}

// ------------------------------ layernorm ----------------------------------
__global__ __launch_bounds__(256) void ln_kernel(
    const float* __restrict__ src, const float* __restrict__ g,
    const float* __restrict__ bb, unsigned short* __restrict__ dst)
{
    __shared__ float r1[256], r2[256];
    const size_t row = blockIdx.x;
    const int t = threadIdx.x;
    float v[4];
    float s1 = 0.f, s2 = 0.f;
#pragma unroll
    for (int i = 0; i < 4; ++i) {
        v[i] = src[row * CC + t * 4 + i];
        s1 += v[i];
        s2 += v[i] * v[i];
    }
    r1[t] = s1; r2[t] = s2;
    __syncthreads();
    for (int s = 128; s > 0; s >>= 1) {
        if (t < s) { r1[t] += r1[t + s]; r2[t] += r2[t + s]; }
        __syncthreads();
    }
    float mean = r1[0] * (1.0f / 1024.0f);
    float var  = r2[0] * (1.0f / 1024.0f) - mean * mean;
    float rstd = rsqrtf(var + 1e-5f);
#pragma unroll
    for (int i = 0; i < 4; ++i) {
        int c = t * 4 + i;
        dst[row * CC + c] = f2bf((v[i] - mean) * rstd * g[c] + bb[c]);
    }
}

// ------------------------------ launch -------------------------------------
extern "C" void kernel_launch(void* const* d_in, const int* in_sizes, int n_in,
                              void* d_out, int out_size, void* d_ws, size_t ws_size,
                              hipStream_t stream)
{
    (void)in_sizes; (void)n_in; (void)out_size; (void)ws_size;
    const float* x   = (const float*)d_in[0];
    const float* Wq  = (const float*)d_in[1];
    const float* bq  = (const float*)d_in[2];
    const float* Wk  = (const float*)d_in[3];
    const float* bk  = (const float*)d_in[4];
    const float* Wv  = (const float*)d_in[5];
    const float* bv  = (const float*)d_in[6];
    const float* Wo  = (const float*)d_in[7];
    const float* bo  = (const float*)d_in[8];
    const float* lng = (const float*)d_in[9];
    const float* lnb = (const float*)d_in[10];
    float* out = (float*)d_out;

    char* ws = (char*)d_ws;
    unsigned short* xb  = (unsigned short*)(ws + OFF_XB);
    unsigned short* qb  = (unsigned short*)(ws + OFF_QB);
    unsigned short* k0b = (unsigned short*)(ws + OFF_K0B);
    unsigned short* k1b = (unsigned short*)(ws + OFF_K1B);
    unsigned short* v0b = (unsigned short*)(ws + OFF_V0B);
    unsigned short* v1b = (unsigned short*)(ws + OFF_V1B);
    unsigned short* wqb = (unsigned short*)(ws + OFF_WQB);
    unsigned short* wkb = (unsigned short*)(ws + OFF_WKB);
    unsigned short* wvb = (unsigned short*)(ws + OFF_WVB);
    unsigned short* wob = (unsigned short*)(ws + OFF_WOB);
    unsigned short* stT = (unsigned short*)(ws + OFF_ST);
    float*          opr = (float*)(ws + OFF_OP);
    unsigned short* xln = (unsigned short*)(ws + OFF_XLN);

    // 1) bf16 converts (weights keep row-major [out][in] == transposed-B layout)
    cvt_bf16_kernel<<<(16384 * 1024 + 255) / 256, 256, 0, stream>>>(x,  xb,  16384 * 1024);
    cvt_bf16_kernel<<<(1024 * 1024 + 255) / 256, 256, 0, stream>>>(Wq, wqb, 1024 * 1024);
    cvt_bf16_kernel<<<(2 * 1024 * 1024 + 255) / 256, 256, 0, stream>>>(Wk, wkb, 2 * 1024 * 1024);
    cvt_bf16_kernel<<<(2 * 1024 * 1024 + 255) / 256, 256, 0, stream>>>(Wv, wvb, 2 * 1024 * 1024);
    cvt_bf16_kernel<<<(1024 * 1024 + 255) / 256, 256, 0, stream>>>(Wo, wob, 1024 * 1024);

    // 2) fused q/k/v GEMMs with epilogues
    dim3 g(128, 8), blk(256);
    const float scale = 0.015625f;  // 4096^-0.5
    gemm_bf16_kernel<<<g, blk, 0, stream>>>(xb, wqb,               x, bq,        qb,  nullptr, 1024, 1024, 0, 1.0f);
    gemm_bf16_kernel<<<g, blk, 0, stream>>>(xb, wkb,               x, bk,        k0b, nullptr, 1024, 1024, 1, scale);
    gemm_bf16_kernel<<<g, blk, 0, stream>>>(xb, wkb + 1024 * 1024, x, bk + 1024, k1b, nullptr, 1024, 1024, 1, scale);
    gemm_bf16_kernel<<<g, blk, 0, stream>>>(xb, wvb,               x, bv,        v0b, nullptr, 1024, 1024, 2, scale);
    gemm_bf16_kernel<<<g, blk, 0, stream>>>(xb, wvb + 1024 * 1024, x, bv + 1024, v1b, nullptr, 1024, 1024, 2, scale);

    // 3) per-head states + elementwise product
    state_kernel<<<64, 256, 0, stream>>>(k0b, k1b, v0b, v1b, stT);

    // 4) out = q @ state (f32 pre-LN)
    out_kernel<<<dim3(32, 64), blk, 0, stream>>>(qb, stT, opr);

    // 5) LayerNorm -> bf16
    ln_kernel<<<16384, 256, 0, stream>>>(opr, lng, lnb, xln);

    // 6) final projection: out = ln(out) @ Wo^T + bo (f32)
    gemm_bf16_kernel<<<g, blk, 0, stream>>>(xln, wob, nullptr, bo, nullptr, out, 1024, 1024, 3, 1.0f);
}
